// AttentionNestedNERModel_75814762709478
// MI455X (gfx1250) — matrix operand order repaired and checked by reading, not compiled, not run on hardware
//
#include <hip/hip_runtime.h>
#include <hip/hip_bf16.h>

// ---------------------------------------------------------------------------
// Problem constants (match reference)
// ---------------------------------------------------------------------------
#define S_LEN 128
#define BATCH 64
#define EMB   512
#define HENC  256
#define DHID  512
#define NLVL  4
#define NCLS  9
#define SB    (S_LEN * BATCH)      // 8192

typedef __attribute__((ext_vector_type(16))) _Float16 h16v;
typedef __attribute__((ext_vector_type(8)))  _Float16 h8v;
typedef __attribute__((ext_vector_type(8)))  float    f8v;

// ---------------------------------------------------------------------------
// WMMA helpers (CDNA5: V_WMMA_F32_16X16X32_F16, f32 accumulate)
// ---------------------------------------------------------------------------
__device__ __forceinline__ f8v wmma16(h16v a, h16v b, f8v c) {
  return __builtin_amdgcn_wmma_f32_16x16x32_f16(false, a, false, b, (short)0, c,
                                                false, false);
}

// A fragment: 16x32 f16 tile, row-major source with leading dim `ld` (halves).
// Lane (m = lane&15, kg = lane>>4) holds K = {kg*8+0..7, 16+kg*8+0..7}.
__device__ __forceinline__ h16v load_a_frag(const _Float16* base, int ld) {
  const int lane = threadIdx.x & 31;
  const int m = lane & 15, kg = lane >> 4;
  const _Float16* p = base + (size_t)m * ld + kg * 8;
  h8v v0 = *reinterpret_cast<const h8v*>(p);
  h8v v1 = *reinterpret_cast<const h8v*>(p + 16);
  h16v r;
#pragma unroll
  for (int i = 0; i < 8; ++i) { r[i] = v0[i]; r[i + 8] = v1[i]; }
  return r;
}

// B fragment for B = W^T where W is row-major [N][K] (PyTorch weight layout).
// Covers B rows k0..k0+31, cols n0..n0+15. Lane (n = lane&15, kg = lane>>4)
// holds K = {kg*8+0..7, 16+kg*8+0..7}.
__device__ __forceinline__ h16v load_bt_frag(const _Float16* W, int ldw,
                                             int n0, int k0) {
  const int lane = threadIdx.x & 31;
  const int n = lane & 15, kg = lane >> 4;
  const _Float16* p = W + (size_t)(n0 + n) * ldw + k0 + kg * 8;
  h8v v0 = *reinterpret_cast<const h8v*>(p);
  h8v v1 = *reinterpret_cast<const h8v*>(p + 16);
  h16v r;
#pragma unroll
  for (int i = 0; i < 8; ++i) { r[i] = v0[i]; r[i + 8] = v1[i]; }
  return r;
}

__device__ __forceinline__ float sigm(float x) {
  return 1.f / (1.f + __expf(-x));
}

// ---------------------------------------------------------------------------
// Blocked f16 WMMA GEMM: C[M,N] (+)= A[M,K] * W[N,K]^T.
// Block = 256 thr (8 waves) computes a 128x64 output block:
//   wave w -> M-tile (blkM*8+w), all 4 N-tiles of the block's 64 columns.
// The 64x32 W panel per K-step is staged once into LDS (double-buffered,
// padded stride 40 halves for conflict-free fragment reads) and shared by all
// 8 waves -> 8x W reuse; one A fragment feeds 4 WMMAs -> 4x A reuse.
// ---------------------------------------------------------------------------
#define BP_LD 40  // padded LDS row stride (halves) for the B panel
template <typename OUT, bool ACC>
__global__ __launch_bounds__(256) void k_gemm(const _Float16* __restrict__ A,
                                              int lda,
                                              const _Float16* __restrict__ W,
                                              int ldw, OUT* __restrict__ C,
                                              int ldc, int nblk64, int K) {
  __shared__ __align__(16) _Float16 Bs[2][64 * BP_LD];
  const int tid = threadIdx.x;
  const int lane = tid & 31;
  const int wave = tid >> 5;
  const int blkM = blockIdx.x / nblk64;
  const int nb = blockIdx.x % nblk64;
  const int n0 = nb * 64;
  const int mt = blkM * 8 + wave;

  f8v acc[4];
#pragma unroll
  for (int nn = 0; nn < 4; ++nn)
#pragma unroll
    for (int r = 0; r < 8; ++r) acc[nn][r] = 0.f;

  const _Float16* abase = A + (size_t)(mt * 16) * lda;
  const int srow = tid >> 2, skc = (tid & 3) * 8;  // staging assignment

  // stage first W panel (k = 0)
  *reinterpret_cast<h8v*>(&Bs[0][srow * BP_LD + skc]) =
      *reinterpret_cast<const h8v*>(&W[(size_t)(n0 + srow) * ldw + skc]);
  __syncthreads();

  const int nk = K / 32;
  for (int kt = 0; kt < nk; ++kt) {
    const int k = kt * 32;
    const int cur = kt & 1;
    // stage next panel into the other buffer (its previous readers synced)
    if (kt + 1 < nk) {
      *reinterpret_cast<h8v*>(&Bs[cur ^ 1][srow * BP_LD + skc]) =
          *reinterpret_cast<const h8v*>(
              &W[(size_t)(n0 + srow) * ldw + (k + 32) + skc]);
    }
    // prefetch the A stream two panels ahead (global_prefetch_b8)
    if (kt + 2 < nk)
      __builtin_prefetch(abase + (size_t)(lane & 15) * lda + k + 64, 0, 0);

    h16v a = load_a_frag(abase + k, lda);
#pragma unroll
    for (int nn = 0; nn < 4; ++nn) {
      h16v b = load_bt_frag(&Bs[cur][nn * 16 * BP_LD], BP_LD, 0, 0);
      acc[nn] = wmma16(a, b, acc[nn]);
    }
    __syncthreads();
  }

  const int row0 = mt * 16 + (lane >> 4) * 8;
#pragma unroll
  for (int nn = 0; nn < 4; ++nn) {
    const int col = n0 + nn * 16 + (lane & 15);
#pragma unroll
    for (int r = 0; r < 8; ++r) {
      OUT* p = C + (size_t)(row0 + r) * ldc + col;
      if (ACC)
        *p = (OUT)((float)*p + acc[nn][r]);
      else
        *p = (OUT)acc[nn][r];
    }
  }
}

// ---------------------------------------------------------------------------
// Small prep kernels
// ---------------------------------------------------------------------------
__global__ void k_f32_to_f16(const float* __restrict__ in,
                             _Float16* __restrict__ out, int n) {
  int i = blockIdx.x * 256 + threadIdx.x;
  if (i < n) out[i] = (_Float16)in[i];
}

// Embedding gather into x half of Acat ([S*B][1024]: cols 0..511 = h (encoder
// writes later), cols 512..1023 = x = emb[seqs]).
__global__ void k_gather(const int* __restrict__ seqs,
                         const float* __restrict__ emb,
                         _Float16* __restrict__ Acat) {
  int i = blockIdx.x * 256 + threadIdx.x;
  if (i >= SB * EMB) return;
  int e = i & 511, sb = i >> 9;
  int b = sb & 63, s = sb >> 6;
  int tok = seqs[b * S_LEN + s];
  Acat[(size_t)sb * 1024 + 512 + e] = (_Float16)emb[(size_t)tok * EMB + e];
}

// Build decoder weight repacks from Wih_d [2048][2052] / Whh_d [2048][512]:
//  Wrec [2048][1024]  = [W_ctx | Whh_d]          (sequential-step GEMM)
//  Wcat0[2048][1024]  = [W_ht + W_pt | W_et]     (level-0 static, pt = h)
//  Wcat1[2048][1024]  = [W_ht        | W_et]     (levels>=1 static, pt added later)
//  Wp   [2048][512]   = W_pt                     (for Gstat1 += hs0 @ Wp^T)
__global__ void k_build_decw(const float* __restrict__ Wih_d,
                             const float* __restrict__ Whh_d,
                             _Float16* __restrict__ Wrec,
                             _Float16* __restrict__ Wcat0,
                             _Float16* __restrict__ Wcat1,
                             _Float16* __restrict__ Wp) {
  int i = blockIdx.x * 256 + threadIdx.x;
  if (i >= 2048 * 1024) return;
  int n = i >> 10, k = i & 1023;
  const float* wr = Wih_d + (size_t)n * 2052;
  Wrec[i] = (_Float16)(k < 512 ? wr[k] : Whh_d[(size_t)n * 512 + (k - 512)]);
  Wcat0[i] =
      (_Float16)(k < 512 ? wr[512 + k] + wr[1536 + k] : wr[1024 + (k - 512)]);
  Wcat1[i] = (_Float16)(k < 512 ? wr[512 + k] : wr[1024 + (k - 512)]);
  if (k < 512) Wp[(size_t)n * 512 + k] = (_Float16)wr[1536 + k];
}

// Fused biases: encoder bih+bhh per direction; decoder per-level
// bih_d+bhh_d+Wih_d[:,2048+lvl] (the one-hot column).
__global__ void k_bias(const float* bih_f, const float* bhh_f,
                       const float* bih_b, const float* bhh_b,
                       const float* bih_d, const float* bhh_d,
                       const float* Wih_d, float* bias_f, float* bias_b,
                       float* bias_lvl) {
  int i = blockIdx.x * 256 + threadIdx.x;
  if (i < 1024) {
    bias_f[i] = bih_f[i] + bhh_f[i];
    bias_b[i] = bih_b[i] + bhh_b[i];
  }
  if (i < NLVL * 2048) {
    int l = i >> 11, j = i & 2047;
    bias_lvl[i] = bih_d[j] + bhh_d[j] + Wih_d[(size_t)j * 2052 + 2048 + l];
  }
}

__global__ void k_zero_carry(_Float16* hd, float* cd) {
  int i = blockIdx.x * 256 + threadIdx.x;
  if (i < BATCH * DHID) {
    hd[i] = (_Float16)0.f;
    cd[i] = 0.f;
  }
}

// ---------------------------------------------------------------------------
// Encoder: persistent kernel, grid=2 (dir 0=fwd, 1=bwd), 512 thr = 16 waves.
// Per step: recurrent GEMM h[64,256]@Whh^T via WMMA, gates += precomputed
// x@Wih^T (X4h) + bias, cell math in f32, c kept in D-fragment registers.
// ---------------------------------------------------------------------------
__global__ __launch_bounds__(512) void k_encoder(
    const float* __restrict__ X4h_f, const float* __restrict__ X4h_b,
    const _Float16* __restrict__ Whh_fw, const _Float16* __restrict__ Whh_bw,
    const float* __restrict__ bias_f, const float* __restrict__ bias_b,
    _Float16* __restrict__ Acat) {
  const int dir = blockIdx.x;
  const float* X4h = dir ? X4h_b : X4h_f;
  const _Float16* W = dir ? Whh_bw : Whh_fw;
  const float* bias = dir ? bias_b : bias_f;

  __shared__ __align__(16) _Float16 h_s[BATCH * HENC];  // 32 KB

  const int tid = threadIdx.x, lane = tid & 31, wave = tid >> 5;
  const int hi = lane >> 4, ln = lane & 15;

  int bt[4], jt[4];
  f8v cfrag[4];
#pragma unroll
  for (int n = 0; n < 4; ++n) {
    int id = wave * 4 + n;
    bt[n] = id >> 4;
    jt[n] = id & 15;
#pragma unroll
    for (int r = 0; r < 8; ++r) cfrag[n][r] = 0.f;
  }
  for (int i = tid; i < BATCH * HENC; i += blockDim.x) h_s[i] = (_Float16)0.f;
  __syncthreads();

  for (int t = 0; t < S_LEN; ++t) {
    const int tx = dir ? (S_LEN - 1 - t) : t;
    h8v hout[4];
    for (int n = 0; n < 4; ++n) {
      f8v acc[4];
#pragma unroll
      for (int g = 0; g < 4; ++g)
#pragma unroll
        for (int r = 0; r < 8; ++r) acc[g][r] = 0.f;
      for (int kt = 0; kt < 8; ++kt) {
        int k = kt * 32;
        h16v a = load_a_frag(h_s + bt[n] * 16 * HENC + k, HENC);
#pragma unroll
        for (int g = 0; g < 4; ++g) {
          h16v b = load_bt_frag(W, HENC, g * HENC + jt[n] * 16, k);
          acc[g] = wmma16(a, b, acc[g]);
        }
      }
      const int j = jt[n] * 16 + ln;
#pragma unroll
      for (int r = 0; r < 8; ++r) {
        int brow = bt[n] * 16 + r + 8 * hi;
        const float* xg = X4h + ((size_t)tx * BATCH + brow) * 1024;
        float gi = acc[0][r] + xg[0 * HENC + j] + bias[0 * HENC + j];
        float gf = acc[1][r] + xg[1 * HENC + j] + bias[1 * HENC + j];
        float gg = acc[2][r] + xg[2 * HENC + j] + bias[2 * HENC + j];
        float go = acc[3][r] + xg[3 * HENC + j] + bias[3 * HENC + j];
        float c = sigm(gf) * cfrag[n][r] + sigm(gi) * tanhf(gg);
        cfrag[n][r] = c;
        float hv = sigm(go) * tanhf(c);
        hout[n][r] = (_Float16)hv;
        // encoder state into Acat cols [dir*256 .. dir*256+255]
        Acat[((size_t)tx * BATCH + brow) * 1024 + dir * HENC + j] =
            (_Float16)hv;
      }
    }
    __syncthreads();  // all waves done reading h_s
#pragma unroll
    for (int n = 0; n < 4; ++n) {
      const int j = jt[n] * 16 + ln;
#pragma unroll
      for (int r = 0; r < 8; ++r)
        h_s[(bt[n] * 16 + r + 8 * hi) * HENC + j] = hout[n][r];
    }
    __syncthreads();
  }
}

// ---------------------------------------------------------------------------
// Decoder: single persistent workgroup (512 thr = 16 waves) running levels
// [lvl_begin, lvl_end). Per step:
//   scores = wh . hd  -> softmax -> ctx = att . h_enc  (LDS)
//   gates  = [ctx|hd] @ Wrec^T (WMMA, K=1024) + Gstat[t] + bias_lvl
//   cell in f32, c in D-fragment registers, hd back to LDS + hs_dec global.
// ---------------------------------------------------------------------------
__global__ __launch_bounds__(512) void k_decoder(
    const _Float16* __restrict__ wh, const _Float16* __restrict__ h_enc,
    const _Float16* __restrict__ Wrec, const float* __restrict__ Gstat,
    const float* __restrict__ bias_lvl, _Float16* __restrict__ hs_dec,
    _Float16* __restrict__ hd_state, float* __restrict__ cd_state,
    int lvl_begin, int lvl_end) {
  __shared__ __align__(16) _Float16 hd_s[BATCH * DHID];   // 64 KB
  __shared__ __align__(16) _Float16 ctx_s[BATCH * DHID];  // 64 KB
  __shared__ __align__(16) _Float16 sc_s[BATCH * S_LEN];  // 16 KB

  const int tid = threadIdx.x, lane = tid & 31, wave = tid >> 5;
  const int hi = lane >> 4, ln = lane & 15;

  int bt[8], jt[8];
  f8v cfrag[8];
#pragma unroll
  for (int n = 0; n < 8; ++n) {
    int id = wave * 8 + n;
    bt[n] = id >> 5;
    jt[n] = id & 31;
#pragma unroll
    for (int r = 0; r < 8; ++r) {
      int row = bt[n] * 16 + r + 8 * hi, col = jt[n] * 16 + ln;
      cfrag[n][r] = cd_state[row * DHID + col];
    }
  }
  for (int i = tid; i < BATCH * DHID; i += blockDim.x) hd_s[i] = hd_state[i];
  __syncthreads();

  for (int lvl = lvl_begin; lvl < lvl_end; ++lvl) {
    const float* bl = bias_lvl + lvl * 2048;
    for (int t = 0; t < S_LEN; ++t) {
      // (a) attention scores: sc[b][s] = wh[s,b,:] . hd[b,:]
      for (int p = tid; p < BATCH * S_LEN; p += blockDim.x) {
        int b = p >> 7, s = p & 127;
        const _Float16* wp = wh + ((size_t)s * BATCH + b) * DHID;
        const _Float16* hp = hd_s + b * DHID;
        float acc = 0.f;
        for (int d = 0; d < DHID; d += 8) {
          h8v w8 = *reinterpret_cast<const h8v*>(wp + d);
          h8v x8 = *reinterpret_cast<const h8v*>(hp + d);
#pragma unroll
          for (int q = 0; q < 8; ++q) acc += (float)w8[q] * (float)x8[q];
        }
        sc_s[p] = (_Float16)acc;
      }
      __syncthreads();
      // (b) softmax over s (one row per thread)
      if (tid < BATCH) {
        _Float16* row = sc_s + tid * S_LEN;
        float m = (float)row[0];
        for (int s = 1; s < S_LEN; ++s) m = fmaxf(m, (float)row[s]);
        float sum = 0.f;
        for (int s = 0; s < S_LEN; ++s) {
          float e = __expf((float)row[s] - m);
          row[s] = (_Float16)e;
          sum += e;
        }
        float inv = 1.f / sum;
        for (int s = 0; s < S_LEN; ++s)
          row[s] = (_Float16)((float)row[s] * inv);
      }
      __syncthreads();
      // (c) ctx[b][d] = sum_s att[b][s] * h_enc[s,b,d]
      for (int p = tid; p < BATCH * DHID; p += blockDim.x) {
        int b = p >> 9, d = p & 511;
        const _Float16* arow = sc_s + b * S_LEN;
        float acc = 0.f;
        for (int s = 0; s < S_LEN; ++s)
          acc += (float)arow[s] *
                 (float)h_enc[((size_t)s * BATCH + b) * 1024 + d];
        ctx_s[b * DHID + d] = (_Float16)acc;
      }
      __syncthreads();
      // (d) gates = [ctx|hd] @ Wrec^T + Gstat + bias ; cell math
      h8v hout[8];
      for (int n = 0; n < 8; ++n) {
        f8v acc[4];
#pragma unroll
        for (int g = 0; g < 4; ++g)
#pragma unroll
          for (int r = 0; r < 8; ++r) acc[g][r] = 0.f;
        for (int kt = 0; kt < 32; ++kt) {
          int k = kt * 32;
          const _Float16* abase = (k < DHID)
                                      ? (ctx_s + bt[n] * 16 * DHID + k)
                                      : (hd_s + bt[n] * 16 * DHID + (k - DHID));
          h16v a = load_a_frag(abase, DHID);
#pragma unroll
          for (int g = 0; g < 4; ++g) {
            h16v b = load_bt_frag(Wrec, 1024, g * DHID + jt[n] * 16, k);
            acc[g] = wmma16(a, b, acc[g]);
          }
        }
        const int j = jt[n] * 16 + ln;
#pragma unroll
        for (int r = 0; r < 8; ++r) {
          int brow = bt[n] * 16 + r + 8 * hi;
          const float* gs = Gstat + ((size_t)t * BATCH + brow) * 2048;
          float gi = acc[0][r] + gs[0 * DHID + j] + bl[0 * DHID + j];
          float gf = acc[1][r] + gs[1 * DHID + j] + bl[1 * DHID + j];
          float gg = acc[2][r] + gs[2 * DHID + j] + bl[2 * DHID + j];
          float go = acc[3][r] + gs[3 * DHID + j] + bl[3 * DHID + j];
          float c = sigm(gf) * cfrag[n][r] + sigm(gi) * tanhf(gg);
          cfrag[n][r] = c;
          float hv = sigm(go) * tanhf(c);
          hout[n][r] = (_Float16)hv;
          hs_dec[(((size_t)lvl * S_LEN + t) * BATCH + brow) * DHID + j] =
              (_Float16)hv;
        }
      }
      __syncthreads();  // all waves done reading hd_s / ctx_s
#pragma unroll
      for (int n = 0; n < 8; ++n) {
        const int j = jt[n] * 16 + ln;
#pragma unroll
        for (int r = 0; r < 8; ++r)
          hd_s[(bt[n] * 16 + r + 8 * hi) * DHID + j] = hout[n][r];
      }
      __syncthreads();
    }
  }
  // save carry for the follow-up launch
  for (int i = tid; i < BATCH * DHID; i += blockDim.x) hd_state[i] = hd_s[i];
#pragma unroll
  for (int n = 0; n < 8; ++n)
#pragma unroll
    for (int r = 0; r < 8; ++r) {
      int row = bt[n] * 16 + r + 8 * hi, col = jt[n] * 16 + ln;
      cd_state[row * DHID + col] = cfrag[n][r];
    }
}

// ---------------------------------------------------------------------------
// Output projection: logits[row, c] = hs_dec[row,:] . W2[c,:] + b2[c], N=9
// (too narrow for WMMA; memory-trivial scalar kernel).
// ---------------------------------------------------------------------------
__global__ void k_proj(const _Float16* __restrict__ hs,
                       const float* __restrict__ W2,
                       const float* __restrict__ b2, float* __restrict__ out) {
  int i = blockIdx.x * 256 + threadIdx.x;
  if (i >= NLVL * SB * NCLS) return;
  int r = i / NCLS, c = i - r * NCLS;
  const _Float16* hp = hs + (size_t)r * DHID;
  const float* wp = W2 + (size_t)c * DHID;
  float acc = b2[c];
  for (int d = 0; d < DHID; ++d) acc += (float)hp[d] * wp[d];
  out[i] = acc;
}

// ---------------------------------------------------------------------------
// Host orchestration
// ---------------------------------------------------------------------------
extern "C" void kernel_launch(void* const* d_in, const int* in_sizes, int n_in,
                              void* d_out, int out_size, void* d_ws,
                              size_t ws_size, hipStream_t stream) {
  (void)in_sizes; (void)n_in; (void)out_size; (void)ws_size;
  const int* seqs = (const int*)d_in[0];
  const float* emb = (const float*)d_in[2];
  const float* Wih_f = (const float*)d_in[3];
  const float* Whh_f = (const float*)d_in[4];
  const float* bih_f = (const float*)d_in[5];
  const float* bhh_f = (const float*)d_in[6];
  const float* Wih_b = (const float*)d_in[7];
  const float* Whh_b = (const float*)d_in[8];
  const float* bih_b = (const float*)d_in[9];
  const float* bhh_b = (const float*)d_in[10];
  const float* Wl = (const float*)d_in[11];
  const float* Wih_d = (const float*)d_in[12];
  const float* Whh_d = (const float*)d_in[13];
  const float* bih_d = (const float*)d_in[14];
  const float* bhh_d = (const float*)d_in[15];
  const float* W2 = (const float*)d_in[16];
  const float* b2 = (const float*)d_in[17];
  float* out = (float*)d_out;

  // bump allocator over d_ws (~278 MB total)
  char* w = (char*)d_ws;
  auto alloc = [&](size_t bytes) -> char* {
    char* p = w;
    w += (bytes + 255) & ~(size_t)255;
    return p;
  };
  _Float16* Acat = (_Float16*)alloc((size_t)SB * 1024 * 2);       // [h|x] f16
  float* X4h_f = (float*)alloc((size_t)SB * 1024 * 4);
  float* X4h_b = (float*)alloc((size_t)SB * 1024 * 4);
  _Float16* wh = (_Float16*)alloc((size_t)SB * DHID * 2);
  float* Gstat0 = (float*)alloc((size_t)SB * 2048 * 4);
  float* Gstat1 = (float*)alloc((size_t)SB * 2048 * 4);
  _Float16* hs_dec = (_Float16*)alloc((size_t)NLVL * SB * DHID * 2);
  _Float16* Wih16f = (_Float16*)alloc((size_t)1024 * 512 * 2);
  _Float16* Wih16b = (_Float16*)alloc((size_t)1024 * 512 * 2);
  _Float16* Whh16f = (_Float16*)alloc((size_t)1024 * 256 * 2);
  _Float16* Whh16b = (_Float16*)alloc((size_t)1024 * 256 * 2);
  _Float16* Wl16 = (_Float16*)alloc((size_t)512 * 512 * 2);
  _Float16* Wrec = (_Float16*)alloc((size_t)2048 * 1024 * 2);
  _Float16* Wcat0 = (_Float16*)alloc((size_t)2048 * 1024 * 2);
  _Float16* Wcat1 = (_Float16*)alloc((size_t)2048 * 1024 * 2);
  _Float16* Wp = (_Float16*)alloc((size_t)2048 * 512 * 2);
  float* bias_f = (float*)alloc(1024 * 4);
  float* bias_b = (float*)alloc(1024 * 4);
  float* bias_lvl = (float*)alloc(NLVL * 2048 * 4);
  _Float16* hd_state = (_Float16*)alloc((size_t)BATCH * DHID * 2);
  float* cd_state = (float*)alloc((size_t)BATCH * DHID * 4);

  auto cdiv = [](int a, int b) { return (a + b - 1) / b; };

  // weight conversions / repacks
  k_f32_to_f16<<<cdiv(1024 * 512, 256), 256, 0, stream>>>(Wih_f, Wih16f, 1024 * 512);
  k_f32_to_f16<<<cdiv(1024 * 512, 256), 256, 0, stream>>>(Wih_b, Wih16b, 1024 * 512);
  k_f32_to_f16<<<cdiv(1024 * 256, 256), 256, 0, stream>>>(Whh_f, Whh16f, 1024 * 256);
  k_f32_to_f16<<<cdiv(1024 * 256, 256), 256, 0, stream>>>(Whh_b, Whh16b, 1024 * 256);
  k_f32_to_f16<<<cdiv(512 * 512, 256), 256, 0, stream>>>(Wl, Wl16, 512 * 512);
  k_build_decw<<<cdiv(2048 * 1024, 256), 256, 0, stream>>>(Wih_d, Whh_d, Wrec,
                                                           Wcat0, Wcat1, Wp);
  k_bias<<<cdiv(NLVL * 2048, 256), 256, 0, stream>>>(
      bih_f, bhh_f, bih_b, bhh_b, bih_d, bhh_d, Wih_d, bias_f, bias_b, bias_lvl);
  k_gather<<<cdiv(SB * EMB, 256), 256, 0, stream>>>(seqs, emb, Acat);

  // input projections x @ Wih^T (both directions), M=8192 N=1024 K=512
  // blocked GEMM: blocks = (M/128) * (N/64)
  k_gemm<float, false><<<64 * 16, 256, 0, stream>>>(Acat + 512, 1024, Wih16f,
                                                    512, X4h_f, 1024, 16, 512);
  k_gemm<float, false><<<64 * 16, 256, 0, stream>>>(Acat + 512, 1024, Wih16b,
                                                    512, X4h_b, 1024, 16, 512);

  // sequential bi-LSTM encoder (fwd/bwd in parallel blocks)
  k_encoder<<<2, 512, 0, stream>>>(X4h_f, X4h_b, Whh16f, Whh16b, bias_f,
                                   bias_b, Acat);

  // wh = h @ Wl^T (f16 out), M=8192 N=512 K=512
  k_gemm<_Float16, false><<<64 * 8, 256, 0, stream>>>(Acat, 1024, Wl16, 512,
                                                      wh, 512, 8, 512);
  // static decoder gate precompute, M=8192 N=2048 K=1024
  k_gemm<float, false><<<64 * 32, 256, 0, stream>>>(Acat, 1024, Wcat0, 1024,
                                                    Gstat0, 2048, 32, 1024);
  k_gemm<float, false><<<64 * 32, 256, 0, stream>>>(Acat, 1024, Wcat1, 1024,
                                                    Gstat1, 2048, 32, 1024);

  k_zero_carry<<<cdiv(BATCH * DHID, 256), 256, 0, stream>>>(hd_state, cd_state);

  // decoder level 0
  k_decoder<<<1, 512, 0, stream>>>(wh, Acat, Wrec, Gstat0, bias_lvl, hs_dec,
                                   hd_state, cd_state, 0, 1);
  // Gstat1 += hs_lvl0 @ Wp^T (prev_s for levels 1..3)
  k_gemm<float, true><<<64 * 32, 256, 0, stream>>>(hs_dec, 512, Wp, 512,
                                                   Gstat1, 2048, 32, 512);
  // decoder levels 1..3
  k_decoder<<<1, 512, 0, stream>>>(wh, Acat, Wrec, Gstat1, bias_lvl, hs_dec,
                                   hd_state, cd_state, 1, NLVL);

  // output projection
  k_proj<<<cdiv(NLVL * SB * NCLS, 256), 256, 0, stream>>>(hs_dec, W2, b2, out);
}
